// _GroundingDinoBiMultiHeadAttention_53343493816659
// MI455X (gfx1250) — compile-verified, matrix operand order checked by reading, not run on hardware
//
#include <hip/hip_runtime.h>
#include <math.h>

// ---------------------------------------------------------------------------
// Types for CDNA5 (gfx1250) WMMA
// ---------------------------------------------------------------------------
typedef __bf16 bf16;
typedef __attribute__((ext_vector_type(16))) bf16  v16bf;
typedef __attribute__((ext_vector_type(8)))  bf16  v8bf;
typedef __attribute__((ext_vector_type(8)))  float v8f;
typedef __attribute__((ext_vector_type(4)))  float v4f;

#define BM 128
#define BN 128
#define BK 64

// ---------------------------------------------------------------------------
// Problem constants (GroundingDino bi-attention)
// ---------------------------------------------------------------------------
static const int  cB  = 2;
static const int  cTV = 16384;
static const int  cTT = 256;
static const int  cVD = 256;
static const int  cTD = 256;
static const int  cE  = 1024;
static const int  cH  = 4;
static const int  cD  = 256;            // head dim
static const float cSCALE = 0.0625f;    // 256^-0.5
static const float cCLAMP = 50000.0f;
static const float cNEGMAX = 3.402823466e38f;

// 8-element vector load (+ bf16 convert when the source is fp32)
__device__ __forceinline__ v8bf load8_as_bf16(const float* __restrict__ p) {
  v4f lo = *(const v4f*)p;
  v4f hi = *(const v4f*)(p + 4);
  v8bf r;
#pragma unroll
  for (int j = 0; j < 4; ++j) { r[j] = (bf16)lo[j]; r[4 + j] = (bf16)hi[j]; }
  return r;
}
__device__ __forceinline__ v8bf load8_as_bf16(const bf16* __restrict__ p) {
  return *(const v8bf*)p;
}

// ---------------------------------------------------------------------------
// Weight pack: fp32 W[K][N] -> bf16 WT[N][K]  (so GEMMs are all "NT")
// ---------------------------------------------------------------------------
__global__ __launch_bounds__(256)
void pack_wt(const float* __restrict__ W, bf16* __restrict__ WT, int K, int N) {
  long i = (long)blockIdx.x * 256 + threadIdx.x;
  if (i < (long)K * N) {
    int k = (int)(i / N), n = (int)(i % N);
    WT[(long)n * K + k] = (bf16)W[i];
  }
}

// ---------------------------------------------------------------------------
// Tiled WMMA GEMM:  C[M,N] = A[M,K] * B[N,K]^T  (+bias)*scale
//   AT      : A element type (float gets converted to bf16 on the LDS stage)
//   OUTMODE : 0 = f32 row-major, 1 = bf16 row-major, 2 = bf16 transposed
// Batched over blockIdx.z with split (z/ZH, z%ZH) strides for (b, h) batching.
// Block tile 128x128x64, 8 waves (wave32), each wave 64x32 -> 16 WMMA / stage.
// ---------------------------------------------------------------------------
template <typename AT, int OUTMODE>
__global__ __launch_bounds__(256)
void gemm_nt(const AT* __restrict__ A, const bf16* __restrict__ Bm,
             const float* __restrict__ bias, float scale, void* __restrict__ Cout,
             int M, int N, int K, int lda, int ldb, int ldc,
             long sAb, long sAh, long sBb, long sBh, long sCb, long sCh, int ZH) {
  __shared__ __align__(16) bf16 sA[BM][BK];
  __shared__ __align__(16) bf16 sB[BN][BK];

  const int  z  = blockIdx.z;
  const int  zb = z / ZH, zh = z % ZH;
  const AT*   Ab = A  + (long)zb * sAb + (long)zh * sAh;
  const bf16* Bb = Bm + (long)zb * sBb + (long)zh * sBh;
  const long  cOff = (long)zb * sCb + (long)zh * sCh;

  const int tid     = threadIdx.x;
  const int rowBase = blockIdx.y * BM;
  const int colBase = blockIdx.x * BN;

  const int wave = tid >> 5;
  const int lane = tid & 31;
  const int wm = (wave >> 2) * 64;    // wave M offset (2 waves in M)
  const int wn = (wave & 3) * 32;     // wave N offset (4 waves in N)
  const int lr = lane & 15;           // fragment row/col within 16
  const int kh = (lane >> 4) * 8;     // K-half select per ISA bf16 layout

  // staging map: 8 contiguous elements per thread per step; 8 groups per row
  const int sr = tid >> 3;            // row 0..31 (+32 per step)
  const int sc = (tid & 7) * 8;       // col 0,8,...,56

  v8f acc[4][2];
#pragma unroll
  for (int mt = 0; mt < 4; ++mt)
#pragma unroll
    for (int nt = 0; nt < 2; ++nt)
#pragma unroll
      for (int i = 0; i < 8; ++i) acc[mt][nt][i] = 0.0f;

  for (int k0 = 0; k0 < K; k0 += BK) {
    // Stage A tile: 128x64, vectorized 16B stores (fp32 src converted here)
#pragma unroll
    for (int it = 0; it < 4; ++it) {
      const int r = sr + it * 32;
      *(v8bf*)&sA[r][sc] = load8_as_bf16(&Ab[(long)(rowBase + r) * lda + (k0 + sc)]);
    }
    // Stage B tile: 128x64 bf16 [N,K] row-major
#pragma unroll
    for (int it = 0; it < 4; ++it) {
      const int r = sr + it * 32;
      *(v8bf*)&sB[r][sc] = *(const v8bf*)&Bb[(long)(colBase + r) * ldb + (k0 + sc)];
    }
    // Prefetch next K-tile into cache while this stage computes
    if (k0 + BK < K) {
      __builtin_prefetch((const void*)&Ab[(long)(rowBase + sr) * lda + (k0 + BK + sc)], 0, 0);
      __builtin_prefetch((const void*)&Bb[(long)(colBase + sr) * ldb + (k0 + BK + sc)], 0, 0);
    }
    __syncthreads();

#pragma unroll
    for (int kk = 0; kk < BK; kk += 32) {
      // Fragments per ISA 16-bit A/B layout:
      // lanes 0-15: K kk+0..7 / kk+16..23 ; lanes 16-31: K kk+8..15 / kk+24..31
      v16bf afr[4], bfr[2];
#pragma unroll
      for (int mt = 0; mt < 4; ++mt) {
        const bf16* p = &sA[wm + mt * 16 + lr][kk];
        v8bf lo = *(const v8bf*)(p + kh);
        v8bf hi = *(const v8bf*)(p + 16 + kh);
#pragma unroll
        for (int i = 0; i < 8; ++i) { afr[mt][i] = lo[i]; afr[mt][i + 8] = hi[i]; }
      }
#pragma unroll
      for (int nt = 0; nt < 2; ++nt) {
        const bf16* p = &sB[wn + nt * 16 + lr][kk];
        v8bf lo = *(const v8bf*)(p + kh);
        v8bf hi = *(const v8bf*)(p + 16 + kh);
#pragma unroll
        for (int i = 0; i < 8; ++i) { bfr[nt][i] = lo[i]; bfr[nt][i + 8] = hi[i]; }
      }
#pragma unroll
      for (int mt = 0; mt < 4; ++mt)
#pragma unroll
        for (int nt = 0; nt < 2; ++nt)
          acc[mt][nt] = __builtin_amdgcn_wmma_f32_16x16x32_bf16(
              false, afr[mt], false, bfr[nt], (short)0, acc[mt][nt], false, false);
    }
    __syncthreads();
  }

  // Epilogue: C layout -> lane holds N=(lane&15), rows M=(lane>>4)*8 + vgpr
  const int rh = (lane >> 4) * 8;
#pragma unroll
  for (int mt = 0; mt < 4; ++mt) {
#pragma unroll
    for (int nt = 0; nt < 2; ++nt) {
      const int col = colBase + wn + nt * 16 + lr;
      const float bv = bias ? bias[col] : 0.0f;
#pragma unroll
      for (int r = 0; r < 8; ++r) {
        const int row = rowBase + wm + mt * 16 + rh + r;
        const float v = (acc[mt][nt][r] + bv) * scale;
        if (OUTMODE == 0)
          ((float*)Cout)[cOff + (long)row * ldc + col] = v;
        else if (OUTMODE == 1)
          ((bf16*)Cout)[cOff + (long)row * ldc + col] = (bf16)v;
        else
          ((bf16*)Cout)[cOff + (long)col * ldc + row] = (bf16)v;
      }
    }
  }
}

// ---------------------------------------------------------------------------
// Global max reduction over scores
// ---------------------------------------------------------------------------
__global__ __launch_bounds__(256)
void reduce_max_part(const float* __restrict__ x, long n, float* __restrict__ part) {
  __shared__ float red[256];
  float m = -cNEGMAX;
  for (long i = (long)blockIdx.x * 256 + threadIdx.x; i < n; i += (long)gridDim.x * 256)
    m = fmaxf(m, x[i]);
  red[threadIdx.x] = m;
  __syncthreads();
  for (int o = 128; o > 0; o >>= 1) {
    if ((int)threadIdx.x < o) red[threadIdx.x] = fmaxf(red[threadIdx.x], red[threadIdx.x + o]);
    __syncthreads();
  }
  if (threadIdx.x == 0) part[blockIdx.x] = red[0];
}

__global__ __launch_bounds__(256)
void reduce_max_final(const float* __restrict__ part, int n, float* __restrict__ g) {
  __shared__ float red[256];
  float m = -cNEGMAX;
  for (int i = threadIdx.x; i < n; i += 256) m = fmaxf(m, part[i]);
  red[threadIdx.x] = m;
  __syncthreads();
  for (int o = 128; o > 0; o >>= 1) {
    if ((int)threadIdx.x < o) red[threadIdx.x] = fmaxf(red[threadIdx.x], red[threadIdx.x + o]);
    __syncthreads();
  }
  if (threadIdx.x == 0) g[0] = red[0];
}

// ---------------------------------------------------------------------------
// vision_w = softmax over TT of clip(scores - gmax) + text-mask(-inf)
// one 256-thread block per (b,h,t) row; TT == 256
// ---------------------------------------------------------------------------
__global__ __launch_bounds__(256)
void vision_softmax(const float* __restrict__ scores, const unsigned char* __restrict__ tmask,
                    const float* __restrict__ gmax, float* __restrict__ out,
                    int TTd, int TVd, int Hd) {
  __shared__ float red[256];
  const int  s   = threadIdx.x;
  const long row = blockIdx.x;                      // (b*H+h)*TV + t
  const int  b   = (int)(row / ((long)Hd * TVd));
  float x = scores[row * TTd + s] - gmax[0];
  x = fminf(fmaxf(x, -cCLAMP), cCLAMP);
  if (!tmask[(long)b * TTd + s]) x = -__builtin_inff();
  red[s] = x;
  __syncthreads();
  for (int o = 128; o > 0; o >>= 1) { if (s < o) red[s] = fmaxf(red[s], red[s + o]); __syncthreads(); }
  const float m = red[0];
  __syncthreads();
  const float e = expf(x - m);
  red[s] = e;
  __syncthreads();
  for (int o = 128; o > 0; o >>= 1) { if (s < o) red[s] += red[s + o]; __syncthreads(); }
  out[row * TTd + s] = e / red[0];
}

// ---------------------------------------------------------------------------
// text_w: per row s (over TT): y = clip(scores[:,s]-gmax); z = clip(y - max(y));
//          softmax over TV of (z + vmask(NEG)); one 256-thread block per row
// ---------------------------------------------------------------------------
__global__ __launch_bounds__(256)
void text_softmax(const float* __restrict__ scores, const unsigned char* __restrict__ vmask,
                  const float* __restrict__ gmax, float* __restrict__ out,
                  int TTd, int TVd, int Hd) {
  __shared__ float red[256];
  const int  tid   = threadIdx.x;
  const long rowid = blockIdx.x;                    // z*TT + s
  const long zi    = rowid / TTd;
  const int  s     = (int)(rowid % TTd);
  const int  b     = (int)(zi / Hd);
  const float* sc  = scores + zi * (long)TVd * TTd;
  const unsigned char* vm = vmask + (long)b * TVd;
  const float g = gmax[0];

  // pass 1: M1 = max_t clip(sc[t,s]-g)
  float m = -cNEGMAX;
  for (int t = tid; t < TVd; t += 256) {
    float y = fminf(fmaxf(sc[(long)t * TTd + s] - g, -cCLAMP), cCLAMP);
    m = fmaxf(m, y);
  }
  red[tid] = m; __syncthreads();
  for (int o = 128; o > 0; o >>= 1) { if (tid < o) red[tid] = fmaxf(red[tid], red[tid + o]); __syncthreads(); }
  const float M1 = red[0]; __syncthreads();

  // pass 2: M2 = max_t (clip(y - M1) + mask)
  float m2 = -cNEGMAX;
  for (int t = tid; t < TVd; t += 256) {
    float y  = fminf(fmaxf(sc[(long)t * TTd + s] - g, -cCLAMP), cCLAMP);
    float zv = fminf(fmaxf(y - M1, -cCLAMP), cCLAMP);
    if (!vm[t]) zv += -cNEGMAX;
    m2 = fmaxf(m2, zv);
  }
  red[tid] = m2; __syncthreads();
  for (int o = 128; o > 0; o >>= 1) { if (tid < o) red[tid] = fmaxf(red[tid], red[tid + o]); __syncthreads(); }
  const float M2 = red[0]; __syncthreads();

  // pass 3: sum of exp
  float sum = 0.0f;
  for (int t = tid; t < TVd; t += 256) {
    float y  = fminf(fmaxf(sc[(long)t * TTd + s] - g, -cCLAMP), cCLAMP);
    float zv = fminf(fmaxf(y - M1, -cCLAMP), cCLAMP);
    if (!vm[t]) zv += -cNEGMAX;
    sum += expf(zv - M2);
  }
  red[tid] = sum; __syncthreads();
  for (int o = 128; o > 0; o >>= 1) { if (tid < o) red[tid] += red[tid + o]; __syncthreads(); }
  const float S = red[0];

  // pass 4: write normalized weights (coalesced along t)
  for (int t = tid; t < TVd; t += 256) {
    float y  = fminf(fmaxf(sc[(long)t * TTd + s] - g, -cCLAMP), cCLAMP);
    float zv = fminf(fmaxf(y - M1, -cCLAMP), cCLAMP);
    if (!vm[t]) zv += -cNEGMAX;
    out[rowid * TVd + t] = expf(zv - M2) / S;
  }
}

// ---------------------------------------------------------------------------
// Host launcher
// ---------------------------------------------------------------------------
extern "C" void kernel_launch(void* const* d_in, const int* in_sizes, int n_in,
                              void* d_out, int out_size, void* d_ws, size_t ws_size,
                              hipStream_t stream) {
  (void)in_sizes; (void)n_in; (void)out_size; (void)ws_size;

  const float* vis   = (const float*)d_in[0];
  const float* txt   = (const float*)d_in[1];
  const unsigned char* vmask = (const unsigned char*)d_in[2];
  const unsigned char* tmask = (const unsigned char*)d_in[3];
  const float* Wq  = (const float*)d_in[4];  const float* bq  = (const float*)d_in[5];
  const float* Wk  = (const float*)d_in[6];  const float* bk  = (const float*)d_in[7];
  const float* Wvv = (const float*)d_in[8];  const float* bvv = (const float*)d_in[9];
  const float* Wvt = (const float*)d_in[10]; const float* bvt = (const float*)d_in[11];
  const float* Wov = (const float*)d_in[12]; const float* bov = (const float*)d_in[13];
  const float* Wot = (const float*)d_in[14]; const float* bot = (const float*)d_in[15];

  // d_out layout: v_out | vision_w | t_out | text_w (flat fp32, return order)
  float* outp = (float*)d_out;
  float* vout = outp;
  float* visw = vout + (size_t)cB * cTV * cVD;
  float* tout = visw + (size_t)cB * cH * cTV * cTT;
  float* texw = tout + (size_t)cB * cTT * cTD;

  // workspace carve-up
  char* w = (char*)d_ws;
  bf16* qb   = (bf16*)w; w += (size_t)cB * cTV * cE * 2;   // [B*TV, E]
  bf16* kb   = (bf16*)w; w += (size_t)cB * cTT * cE * 2;   // [B*TT, E]
  bf16* vvT  = (bf16*)w; w += (size_t)cB * cE * cTV * 2;   // [B, E, TV]
  bf16* tvT  = (bf16*)w; w += (size_t)cB * cE * cTT * 2;   // [B, E, TT]
  bf16* voh  = (bf16*)w; w += (size_t)cB * cTV * cE * 2;   // [B*TV, E]
  bf16* toh  = (bf16*)w; w += (size_t)cB * cTT * cE * 2;   // [B*TT, E]
  bf16* WqT  = (bf16*)w; w += (size_t)cVD * cE * 2;        // [E, VD]
  bf16* WkT  = (bf16*)w; w += (size_t)cTD * cE * 2;        // [E, TD]
  bf16* WvvT = (bf16*)w; w += (size_t)cVD * cE * 2;        // [E, VD]
  bf16* WvtT = (bf16*)w; w += (size_t)cTD * cE * 2;        // [E, TD]
  bf16* WovT = (bf16*)w; w += (size_t)cE * cVD * 2;        // [VD, E]
  bf16* WotT = (bf16*)w; w += (size_t)cE * cTD * 2;        // [TD, E]
  float* scores = (float*)w; w += (size_t)cB * cH * cTV * cTT * 4;  // [B*H, TV, TT]
  float* part  = (float*)w;                                 // 1024 partials + gmax
  float* gmax  = part + 1024;

  // ---- 1. pack weights (transpose to [N,K] bf16) ----
  {
    long n1 = (long)cVD * cE;
    pack_wt<<<dim3((unsigned)((n1 + 255) / 256)), 256, 0, stream>>>(Wq,  WqT,  cVD, cE);
    pack_wt<<<dim3((unsigned)((n1 + 255) / 256)), 256, 0, stream>>>(Wk,  WkT,  cTD, cE);
    pack_wt<<<dim3((unsigned)((n1 + 255) / 256)), 256, 0, stream>>>(Wvv, WvvT, cVD, cE);
    pack_wt<<<dim3((unsigned)((n1 + 255) / 256)), 256, 0, stream>>>(Wvt, WvtT, cTD, cE);
    pack_wt<<<dim3((unsigned)((n1 + 255) / 256)), 256, 0, stream>>>(Wov, WovT, cE, cVD);
    pack_wt<<<dim3((unsigned)((n1 + 255) / 256)), 256, 0, stream>>>(Wot, WotT, cE, cTD);
  }

  // ---- 2. projections ----
  // q = (vis @ Wq + bq) * SCALE  -> bf16 [B*TV, E]
  gemm_nt<float, 1><<<dim3(cE / BN, (cB * cTV) / BM, 1), 256, 0, stream>>>(
      vis, WqT, bq, cSCALE, qb, cB * cTV, cE, cVD, cVD, cVD, cE,
      0, 0, 0, 0, 0, 0, 1);
  // k = txt @ Wk + bk -> bf16 [B*TT, E]
  gemm_nt<float, 1><<<dim3(cE / BN, (cB * cTT) / BM, 1), 256, 0, stream>>>(
      txt, WkT, bk, 1.0f, kb, cB * cTT, cE, cTD, cTD, cTD, cE,
      0, 0, 0, 0, 0, 0, 1);
  // vv = vis @ Wvv + bvv -> bf16 transposed per batch: vvT[b][n][t], ld=TV
  gemm_nt<float, 2><<<dim3(cE / BN, cTV / BM, cB), 256, 0, stream>>>(
      vis, WvvT, bvv, 1.0f, vvT, cTV, cE, cVD, cVD, cVD, cTV,
      (long)cTV * cVD, 0, 0, 0, (long)cE * cTV, 0, 1);
  // tv = txt @ Wvt + bvt -> bf16 transposed: tvT[b][n][s], ld=TT
  gemm_nt<float, 2><<<dim3(cE / BN, cTT / BM, cB), 256, 0, stream>>>(
      txt, WvtT, bvt, 1.0f, tvT, cTT, cE, cTD, cTD, cTD, cTT,
      (long)cTT * cTD, 0, 0, 0, (long)cE * cTT, 0, 1);

  // ---- 3. scores[b,h] = q_head @ k_head^T -> fp32 [B*H, TV, TT] ----
  gemm_nt<bf16, 0><<<dim3(cTT / BN, cTV / BM, cB * cH), 256, 0, stream>>>(
      qb, kb, nullptr, 1.0f, scores, cTV, cTT, cD, cE, cE, cTT,
      (long)cTV * cE, (long)cD,           // A: +b*TV*E, +h*D
      (long)cTT * cE, (long)cD,           // B: +b*TT*E, +h*D
      (long)cH * cTV * cTT, (long)cTV * cTT, cH);

  // ---- 4. global max of scores ----
  reduce_max_part<<<1024, 256, 0, stream>>>(scores, (long)cB * cH * cTV * cTT, part);
  reduce_max_final<<<1, 256, 0, stream>>>(part, 1024, gmax);

  // ---- 5. softmaxes ----
  vision_softmax<<<(unsigned)(cB * cH * cTV), 256, 0, stream>>>(
      scores, tmask, gmax, visw, cTT, cTV, cH);
  text_softmax<<<(unsigned)(cB * cH * cTT), 256, 0, stream>>>(
      scores, vmask, gmax, texw, cTT, cTV, cH);

  // ---- 6. attention-value GEMMs ----
  // v_out heads: vision_w[b,h] (fp32, [TV,TT]) @ tvT[b,h] ([D,TT]^T) -> voh[b*TV, h*D..]
  gemm_nt<float, 1><<<dim3(cD / BN, cTV / BM, cB * cH), 256, 0, stream>>>(
      visw, tvT, nullptr, 1.0f, voh, cTV, cD, cTT, cTT, cTT, cE,
      (long)cH * cTV * cTT, (long)cTV * cTT,
      (long)cE * cTT, (long)cD * cTT,
      (long)cTV * cE, (long)cD, cH);
  // t_out heads: text_w[b,h] (fp32, [TT,TV]) @ vvT[b,h] ([D,TV]^T) -> toh[b*TT, h*D..]
  gemm_nt<float, 1><<<dim3(cD / BN, cTT / BM, cB * cH), 256, 0, stream>>>(
      texw, vvT, nullptr, 1.0f, toh, cTT, cD, cTV, cTV, cTV, cE,
      (long)cH * cTT * cTV, (long)cTT * cTV,
      (long)cE * cTV, (long)cD * cTV,
      (long)cTT * cE, (long)cD, cH);

  // ---- 7. output projections (fp32 out to d_out) ----
  gemm_nt<bf16, 0><<<dim3(cVD / BN, (cB * cTV) / BM, 1), 256, 0, stream>>>(
      voh, WovT, bov, 1.0f, vout, cB * cTV, cVD, cE, cE, cE, cVD,
      0, 0, 0, 0, 0, 0, 1);
  gemm_nt<bf16, 0><<<dim3(cTD / BN, (cB * cTT) / BM, 1), 256, 0, stream>>>(
      toh, WotT, bot, 1.0f, tout, cB * cTT, cTD, cE, cE, cE, cTD,
      0, 0, 0, 0, 0, 0, 1);
}
